// attn_90280212562007
// MI455X (gfx1250) — compile-verified, matrix operand order checked by reading, not run on hardware
//
#include <hip/hip_runtime.h>
#include <hip/hip_bf16.h>

// ---------------------------------------------------------------------------
// CDNA5 (gfx1250, wave32) implementation of the cross-attention block.
// All GEMM-like math goes through v_wmma_f32_16x16x32_f16.
// ---------------------------------------------------------------------------

typedef _Float16 v16h __attribute__((ext_vector_type(16)));
typedef _Float16 h8   __attribute__((ext_vector_type(8)));
typedef float    v8f  __attribute__((ext_vector_type(8)));

// Fragment loader: the 16-bit A/B WMMA layout wants, per lane,
//   halves 0..7  = K {base..base+7}
//   halves 8..15 = K {base+16..base+23}
// where base = (lane>=16 ? 8 : 0) (+ k-step offset). Both runs are contiguous
// 16-byte chunks in a row-major LDS tile, so two ds_load_b128 suffice.
__device__ __forceinline__ v16h ldfrag16(const _Float16* p) {
    h8 lo = *(const h8*)p;
    h8 hi = *(const h8*)(p + 16);
    return __builtin_shufflevector(lo, hi, 0,1,2,3,4,5,6,7,8,9,10,11,12,13,14,15);
}

__device__ __forceinline__ v8f wmma16(v16h a, v16h b, v8f c) {
    return __builtin_amdgcn_wmma_f32_16x16x32_f16(
        /*neg_a=*/false, a, /*neg_b=*/false, b,
        /*c_mod=*/(short)0, c, /*reuse_a=*/false, /*reuse_b=*/false);
}

// ---------------------------------------------------------------------------
// Weight prep: f32 [K x N] -> f16 transposed [N x K].  One-time cost; lets the
// GEMM stage its B tile with contiguous b128 loads/stores (no LDS scatter).
// ---------------------------------------------------------------------------
__global__ __launch_bounds__(256) void cvt_tr_f32_f16(const float* __restrict__ in,
                                                      _Float16* __restrict__ out,
                                                      int K, int N) {
    int i = blockIdx.x * 256 + threadIdx.x;
    if (i < K * N) {
        int k = i / N, n = i - k * N;
        out[(long)n * K + k] = (_Float16)in[i];
    }
}

// ---------------------------------------------------------------------------
// LayerNorm over rows of F=512: one block (256 threads) per row, f16 output.
// ---------------------------------------------------------------------------
__global__ __launch_bounds__(256) void layernorm_f16(const float* __restrict__ x,
                                                     const float* __restrict__ g,
                                                     const float* __restrict__ b,
                                                     _Float16* __restrict__ out) {
    __shared__ float red[256];
    const int F = 512;
    int tid = threadIdx.x;
    long row = blockIdx.x;
    const float* xr = x + row * F;
    float v0 = xr[tid], v1 = xr[tid + 256];

    red[tid] = v0 + v1;
    __syncthreads();
    #pragma unroll
    for (int off = 128; off > 0; off >>= 1) {
        if (tid < off) red[tid] += red[tid + off];
        __syncthreads();
    }
    float mu = red[0] * (1.0f / F);
    __syncthreads();

    float d0 = v0 - mu, d1 = v1 - mu;
    red[tid] = d0 * d0 + d1 * d1;
    __syncthreads();
    #pragma unroll
    for (int off = 128; off > 0; off >>= 1) {
        if (tid < off) red[tid] += red[tid + off];
        __syncthreads();
    }
    float rs = rsqrtf(red[0] * (1.0f / F) + 1e-6f);

    out[row * F + tid]       = (_Float16)(d0 * rs * g[tid] + b[tid]);
    out[row * F + tid + 256] = (_Float16)(d1 * rs * g[tid + 256] + b[tid + 256]);
}

// ---------------------------------------------------------------------------
// WMMA GEMM: C[M,N] = act((A_f16[MxK] @ W + bias) * scale) [+ res]
// W is supplied PRE-TRANSPOSED as BT[N x K] f16 row-major.
// Block tile 128x64, K-step 32, 8 waves, each a 2x2 grid of 16x16 WMMA.
// LDS staging is double-buffered (one barrier per K-step); both A and B tiles
// stage with contiguous b128 loads/stores.  Epilogue specialized by template.
// ---------------------------------------------------------------------------
#define GM 128
#define GN 64
#define GK 32
#define LDA 40   // halves (32 + 8 pad), 80B rows -> 16B aligned chunks
#define LDBT 40

template<bool RELU, bool HAS_RES, bool OUT_F32, bool OUT_F16>
__global__ __launch_bounds__(256)
void gemm_wmma_f16(const _Float16* __restrict__ A, const _Float16* __restrict__ BT,
                   const float* __restrict__ bias, const float* __restrict__ res,
                   float* __restrict__ outF, _Float16* __restrict__ outH,
                   int N, int K, float scale) {
    __shared__ __align__(16) _Float16 As[2][GM * LDA];
    __shared__ __align__(16) _Float16 BsT[2][GN * LDBT];

    int tid  = threadIdx.x;
    int lane = tid & 31, wave = tid >> 5;
    int l16  = lane & 15, hi = lane >> 4;
    int waveM = wave & 3, waveN = wave >> 2;       // 4x2 wave grid
    long blockRow = (long)blockIdx.y * GM;
    int  blockCol = blockIdx.x * GN;

    // Staging index precompute (constant per thread).
    int sa_r0 = tid >> 2,          sa_c0 = (tid & 3) * 8;       // A chunk 0
    int sa_r1 = (tid + 256) >> 2,  sa_c1 = sa_c0;               // A chunk 1
    int sb_n  = tid >> 2,          sb_k  = (tid & 3) * 8;       // B chunk

    const _Float16* Arow0 = A  + (blockRow + sa_r0) * (long)K + sa_c0;
    const _Float16* Arow1 = A  + (blockRow + sa_r1) * (long)K + sa_c1;
    const _Float16* Brow  = BT + (long)(blockCol + sb_n) * K + sb_k;

    auto stage = [&](int buf, int kb) {
        *(uint4*)&As[buf][sa_r0 * LDA + sa_c0] = *(const uint4*)(Arow0 + kb);
        *(uint4*)&As[buf][sa_r1 * LDA + sa_c1] = *(const uint4*)(Arow1 + kb);
        *(uint4*)&BsT[buf][sb_n * LDBT + sb_k] = *(const uint4*)(Brow + kb);
    };

    v8f acc[2][2] = {};

    stage(0, 0);
    __syncthreads();

    int nk = K / GK;
    for (int t = 0; t < nk; ++t) {
        int buf = t & 1;
        if (t + 1 < nk) stage(buf ^ 1, (t + 1) * GK);
        if (t + 2 < nk) {   // prefetch 2 tiles ahead into cache
            int kb2 = (t + 2) * GK;
            __builtin_prefetch(A + (blockRow + (tid >> 1)) * (long)K + kb2, 0, 0);
            __builtin_prefetch(Brow + kb2, 0, 0);
        }

        int koff = hi * 8;
        v16h af[2], bf[2];
        #pragma unroll
        for (int f = 0; f < 2; ++f) {
            int m = waveM * 32 + f * 16 + l16;
            af[f] = ldfrag16(&As[buf][m * LDA + koff]);
            int n = waveN * 32 + f * 16 + l16;
            bf[f] = ldfrag16(&BsT[buf][n * LDBT + koff]);
        }
        #pragma unroll
        for (int fm = 0; fm < 2; ++fm)
            #pragma unroll
            for (int fn = 0; fn < 2; ++fn)
                acc[fm][fn] = wmma16(af[fm], bf[fn], acc[fm][fn]);
        __syncthreads();
    }

    // Epilogue: C layout = VGPR r -> (M = tileM + hi*8 + r, N = tileN + l16).
    #pragma unroll
    for (int fm = 0; fm < 2; ++fm) {
        #pragma unroll
        for (int fn = 0; fn < 2; ++fn) {
            long row0 = blockRow + waveM * 32 + fm * 16 + hi * 8;
            int  col  = blockCol + waveN * 32 + fn * 16 + l16;
            float bcol = bias[col];
            #pragma unroll
            for (int r = 0; r < 8; ++r) {
                float v = (acc[fm][fn][r] + bcol) * scale;
                if (RELU) v = fmaxf(v, 0.0f);
                long idx = (row0 + r) * N + col;
                if (HAS_RES) v += res[idx];
                if (OUT_F32) outF[idx] = v;
                if (OUT_F16) outH[idx] = (_Float16)v;
            }
        }
    }
}

// ---------------------------------------------------------------------------
// Flash-attention (cross): per block = one 64-row Q tile of one (batch, head).
// Q/K/V are f16 [8192 x 512] with head h in columns h*64..h*64+63.
// S = Q K^T via WMMA (Ks row-major == transposed B-fragment layout),
// online softmax in LDS, O accumulated in registers (2 fixed tiles per wave),
// P@V via WMMA with V stored transposed in LDS.  Q is pre-scaled by 1/8.
// ---------------------------------------------------------------------------
#define ALDH 72   // half stride for 64-wide f16 tiles (144B rows)
#define ALDS 66   // float stride for S

__global__ __launch_bounds__(256)
void attn_flash_wmma(const _Float16* __restrict__ Q, const _Float16* __restrict__ Km,
                     const _Float16* __restrict__ Vm, _Float16* __restrict__ Out) {
    __shared__ __align__(16) _Float16 Qs[64 * ALDH];
    __shared__ __align__(16) _Float16 Ks[64 * ALDH];
    __shared__ __align__(16) _Float16 VsT[64 * ALDH];
    __shared__ __align__(16) _Float16 Ps[64 * ALDH];
    __shared__ float Ss[64 * ALDS];
    __shared__ float mrow[64], lrow[64], arow[64];

    int tid  = threadIdx.x;
    int lane = tid & 31, wave = tid >> 5;
    int l16  = lane & 15, hi = lane >> 4;
    int qt = blockIdx.x, head = blockIdx.y, bz = blockIdx.z;
    long tokQ = (long)bz * 2048 + qt * 64;
    int  colH = head * 64;

    // Each wave owns two fixed 16x16 output tiles for the whole kt loop.
    int tile0 = wave * 2;
    int tm0 = ((tile0)     >> 2) * 16, tn0 = ((tile0)     & 3) * 16;
    int tm1 = ((tile0 + 1) >> 2) * 16, tn1 = ((tile0 + 1) & 3) * 16;
    v8f oacc0 = {}, oacc1 = {};

    // Load the Q tile (64x64 halves = 512 uint4 chunks).
    #pragma unroll
    for (int i = 0; i < 2; ++i) {
        int c  = tid + i * 256;
        int r  = c >> 3;
        int c8 = (c & 7) * 8;
        *(uint4*)&Qs[r * ALDH + c8] =
            *(const uint4*)(Q + (tokQ + r) * 512 + colH + c8);
    }
    if (tid < 64) { mrow[tid] = -3.0e38f; lrow[tid] = 0.0f; }
    __syncthreads();

    for (int kt = 0; kt < 32; ++kt) {
        long tokK = (long)bz * 2048 + kt * 64;
        #pragma unroll
        for (int i = 0; i < 2; ++i) {
            int c  = tid + i * 256;
            int r  = c >> 3;
            int c8 = (c & 7) * 8;
            *(uint4*)&Ks[r * ALDH + c8] =
                *(const uint4*)(Km + (tokK + r) * 512 + colH + c8);
            uint4 v = *(const uint4*)(Vm + (tokK + r) * 512 + colH + c8);
            const _Float16* hv = (const _Float16*)&v;
            #pragma unroll
            for (int j = 0; j < 8; ++j) VsT[(c8 + j) * ALDH + r] = hv[j];  // [d][key]
        }
        if (kt + 1 < 32) {  // prefetch next K/V tiles (one 128B row per 4 threads)
            long tokN = tokK + 64;
            __builtin_prefetch(Km + (tokN + (tid & 63)) * 512 + colH, 0, 0);
            __builtin_prefetch(Vm + (tokN + (tid & 63)) * 512 + colH, 0, 0);
        }
        __syncthreads();

        // S = Q K^T : 16 tiles of 16x16, 2 per wave.
        #pragma unroll
        for (int t = 0; t < 2; ++t) {
            int tile = wave * 2 + t;
            int tm = (tile >> 2) * 16, tn = (tile & 3) * 16;
            v8f c = {};
            #pragma unroll
            for (int ks = 0; ks < 64; ks += 32) {
                int ko = ks + hi * 8;
                v16h a = ldfrag16(&Qs[(tm + l16) * ALDH + ko]);
                v16h b = ldfrag16(&Ks[(tn + l16) * ALDH + ko]);  // B[k=d][n=key]=K[key][d]
                c = wmma16(a, b, c);
            }
            #pragma unroll
            for (int r = 0; r < 8; ++r)
                Ss[(tm + hi * 8 + r) * ALDS + tn + l16] = c[r];
        }
        __syncthreads();

        // Online softmax: one thread per q-row.
        if (tid < 64) {
            float m_old = mrow[tid];
            float mx = m_old;
            const float* srow = &Ss[tid * ALDS];
            #pragma unroll 8
            for (int c = 0; c < 64; ++c) mx = fmaxf(mx, srow[c]);
            float alpha = __expf(m_old - mx);
            float lsum = 0.0f;
            _Float16* prow = &Ps[tid * ALDH];
            #pragma unroll 8
            for (int c = 0; c < 64; ++c) {
                float p = __expf(srow[c] - mx);
                prow[c] = (_Float16)p;
                lsum += p;
            }
            mrow[tid] = mx;
            lrow[tid] = lrow[tid] * alpha + lsum;
            arow[tid] = alpha;
        }
        __syncthreads();

        // Rescale register O by alpha, then O += P @ V.
        #pragma unroll
        for (int r = 0; r < 8; ++r) {
            oacc0[r] *= arow[tm0 + hi * 8 + r];
            oacc1[r] *= arow[tm1 + hi * 8 + r];
        }
        #pragma unroll
        for (int ks = 0; ks < 64; ks += 32) {
            int ko = ks + hi * 8;
            v16h a0 = ldfrag16(&Ps[(tm0 + l16) * ALDH + ko]);
            v16h b0 = ldfrag16(&VsT[(tn0 + l16) * ALDH + ko]);   // B[k=key][n=d]=V[key][d]
            oacc0 = wmma16(a0, b0, oacc0);
            v16h a1 = ldfrag16(&Ps[(tm1 + l16) * ALDH + ko]);
            v16h b1 = ldfrag16(&VsT[(tn1 + l16) * ALDH + ko]);
            oacc1 = wmma16(a1, b1, oacc1);
        }
        __syncthreads();
    }

    // Normalize by l and store f16.
    #pragma unroll
    for (int r = 0; r < 8; ++r) {
        int row0 = tm0 + hi * 8 + r;
        Out[(tokQ + row0) * 512 + colH + tn0 + l16] = (_Float16)(oacc0[r] / lrow[row0]);
        int row1 = tm1 + hi * 8 + r;
        Out[(tokQ + row1) * 512 + colH + tn1 + l16] = (_Float16)(oacc1[r] / lrow[row1]);
    }
}

// ---------------------------------------------------------------------------
// Host orchestration
// ---------------------------------------------------------------------------
extern "C" void kernel_launch(void* const* d_in, const int* in_sizes, int n_in,
                              void* d_out, int out_size, void* d_ws, size_t ws_size,
                              hipStream_t stream) {
    (void)in_sizes; (void)n_in; (void)out_size; (void)ws_size;

    const float* tm1   = (const float*)d_in[0];
    const float* tm2   = (const float*)d_in[1];
    const float* ln1_g = (const float*)d_in[2];
    const float* ln1_b = (const float*)d_in[3];
    const float* ln2_g = (const float*)d_in[4];
    const float* ln2_b = (const float*)d_in[5];
    const float* Wq  = (const float*)d_in[6];   const float* bq  = (const float*)d_in[7];
    const float* Wk  = (const float*)d_in[8];   const float* bk  = (const float*)d_in[9];
    const float* Wv  = (const float*)d_in[10];  const float* bv  = (const float*)d_in[11];
    const float* Wfc = (const float*)d_in[12];  const float* bfc = (const float*)d_in[13];
    const float* ff1_ln_g = (const float*)d_in[14]; const float* ff1_ln_b = (const float*)d_in[15];
    const float* ff1_W1 = (const float*)d_in[16];   const float* ff1_b1 = (const float*)d_in[17];
    const float* ff1_W2 = (const float*)d_in[18];   const float* ff1_b2 = (const float*)d_in[19];
    const float* ff2_ln_g = (const float*)d_in[20]; const float* ff2_ln_b = (const float*)d_in[21];
    const float* ff2_W1 = (const float*)d_in[22];   const float* ff2_b1 = (const float*)d_in[23];
    const float* ff2_W2 = (const float*)d_in[24];   const float* ff2_b2 = (const float*)d_in[25];

    const int TOK = 4 * 2048;           // 8192 token rows
    const int F   = 512;
    const long OUT1 = (long)TOK * F;    // elements per output tensor

    // Workspace carve-up.
    char* ws = (char*)d_ws;
    size_t off = 0;
    auto carve = [&](size_t bytes) -> void* {
        void* p = ws + off;
        off += (bytes + 255) & ~(size_t)255;
        return p;
    };
    const size_t WBYTES = (size_t)F * F * sizeof(_Float16);
    const size_t TBYTES = (size_t)TOK * F * sizeof(_Float16);
    const size_t XBYTES = (size_t)TOK * F * sizeof(float);

    _Float16* wh[8];   // pre-transposed f16 weights [N x K]
    const float* wsrc[8] = {Wq, Wk, Wv, Wfc, ff1_W1, ff1_W2, ff2_W1, ff2_W2};
    for (int i = 0; i < 8; ++i) wh[i] = (_Float16*)carve(WBYTES);
    _Float16* t1h = (_Float16*)carve(TBYTES);
    _Float16* t2h = (_Float16*)carve(TBYTES);
    _Float16* q1h = (_Float16*)carve(TBYTES);
    _Float16* k1h = (_Float16*)carve(TBYTES);
    _Float16* v1h = (_Float16*)carve(TBYTES);
    _Float16* q2h = (_Float16*)carve(TBYTES);
    _Float16* k2h = (_Float16*)carve(TBYTES);
    _Float16* v2h = (_Float16*)carve(TBYTES);
    float* x1f = (float*)carve(XBYTES);
    float* x2f = (float*)carve(XBYTES);
    // Reuse: attention outputs overwrite t1h/t2h; FF intermediates reuse q/k bufs.
    _Float16* f12h = t1h;
    _Float16* f21h = t2h;
    _Float16* hh1 = q1h; _Float16* uu1 = k1h;
    _Float16* hh2 = q2h; _Float16* uu2 = k2h;

    float* out1 = (float*)d_out;
    float* out2 = (float*)d_out + OUT1;

    // 1) Convert + transpose weights to f16 [N x K].
    for (int i = 0; i < 8; ++i)
        cvt_tr_f32_f16<<<(F * F + 255) / 256, 256, 0, stream>>>(wsrc[i], wh[i], F, F);

    // 2) LayerNorm inputs -> f16.
    layernorm_f16<<<TOK, 256, 0, stream>>>(tm1, ln1_g, ln1_b, t1h);
    layernorm_f16<<<TOK, 256, 0, stream>>>(tm2, ln2_g, ln2_b, t2h);

    // 3) QKV projections (Q pre-scaled by 1/sqrt(64)).
    dim3 gg(F / GN, TOK / GM);
    gemm_wmma_f16<false,false,false,true><<<gg, 256, 0, stream>>>(t1h, wh[0], bq, nullptr, nullptr, q1h, F, F, 0.125f);
    gemm_wmma_f16<false,false,false,true><<<gg, 256, 0, stream>>>(t1h, wh[1], bk, nullptr, nullptr, k1h, F, F, 1.0f);
    gemm_wmma_f16<false,false,false,true><<<gg, 256, 0, stream>>>(t1h, wh[2], bv, nullptr, nullptr, v1h, F, F, 1.0f);
    gemm_wmma_f16<false,false,false,true><<<gg, 256, 0, stream>>>(t2h, wh[0], bq, nullptr, nullptr, q2h, F, F, 0.125f);
    gemm_wmma_f16<false,false,false,true><<<gg, 256, 0, stream>>>(t2h, wh[1], bk, nullptr, nullptr, k2h, F, F, 1.0f);
    gemm_wmma_f16<false,false,false,true><<<gg, 256, 0, stream>>>(t2h, wh[2], bv, nullptr, nullptr, v2h, F, F, 1.0f);

    // 4) Cross attention (overwrites t1h/t2h with head-concat outputs).
    dim3 ag(2048 / 64, 8, 4);
    attn_flash_wmma<<<ag, 256, 0, stream>>>(q1h, k2h, v2h, f12h);
    attn_flash_wmma<<<ag, 256, 0, stream>>>(q2h, k1h, v1h, f21h);

    // 5) Output projection + residual -> x (f32).
    gemm_wmma_f16<false,true,true,false><<<gg, 256, 0, stream>>>(f12h, wh[3], bfc, tm1, x1f, nullptr, F, F, 1.0f);
    gemm_wmma_f16<false,true,true,false><<<gg, 256, 0, stream>>>(f21h, wh[3], bfc, tm2, x2f, nullptr, F, F, 1.0f);

    // 6) FF block stream 1: out1 = x1 + W2(relu(W1 ln(x1)+b1))+b2
    layernorm_f16<<<TOK, 256, 0, stream>>>(x1f, ff1_ln_g, ff1_ln_b, hh1);
    gemm_wmma_f16<true,false,false,true><<<gg, 256, 0, stream>>>(hh1, wh[4], ff1_b1, nullptr, nullptr, uu1, F, F, 1.0f);
    gemm_wmma_f16<false,true,true,false><<<gg, 256, 0, stream>>>(uu1, wh[5], ff1_b2, x1f, out1, nullptr, F, F, 1.0f);

    // 7) FF block stream 2.
    layernorm_f16<<<TOK, 256, 0, stream>>>(x2f, ff2_ln_g, ff2_ln_b, hh2);
    gemm_wmma_f16<true,false,false,true><<<gg, 256, 0, stream>>>(hh2, wh[6], ff2_b1, nullptr, nullptr, uu2, F, F, 1.0f);
    gemm_wmma_f16<false,true,true,false><<<gg, 256, 0, stream>>>(uu2, wh[7], ff2_b2, x2f, out2, nullptr, F, F, 1.0f);
}
